// Graph_Attention_Layer_515396076338
// MI455X (gfx1250) — matrix operand order repaired and checked by reading, not compiled, not run on hardware
//
#include <hip/hip_runtime.h>
#include <hip/hip_bf16.h>
#include <stdint.h>

typedef __attribute__((ext_vector_type(16))) __bf16 v16bf;
typedef __attribute__((ext_vector_type(8)))  float  v8f;
typedef __attribute__((ext_vector_type(8)))  unsigned int v8u;

#define TILE_G 32   // rows per workgroup in GatedMLP kernel
#define TILE_S 64   // rows per workgroup in segment-softmax kernel

// ---------- small helpers ----------
__device__ inline unsigned short f2bf(float x) {
  unsigned int u = __float_as_uint(x);
  unsigned int r = u + 0x7FFFu + ((u >> 16) & 1u);   // round-to-nearest-even
  return (unsigned short)(r >> 16);
}
__device__ inline float bf2f(unsigned short h) {
  return __uint_as_float(((unsigned int)h) << 16);
}
__device__ inline float sigf(float x) { return 1.0f / (1.0f + __expf(-x)); }
__device__ inline v8f zero8() { v8f z = {}; return z; }

// float atomic max via signed/unsigned integer atomics (works across signs)
__device__ inline void atomicMaxF(float* addr, float val) {
  if (val >= 0.0f) atomicMax((int*)addr, __float_as_int(val));
  else             atomicMin((unsigned int*)addr, __float_as_uint(val));
}

// ---------- WMMA fragment loaders (contiguous-K order, same for A and B, wave32) ----------
// A 16x32 bf16: lane holds row M = mbase + lane%16; halves 0-15 = K k0..k0+15
// with k0 = kbase + (lane/16)*16.
__device__ inline v16bf frag_a_lds(const unsigned short* sh, int stride, int mbase, int kbase) {
  int lane = threadIdx.x & 31;
  const unsigned short* p = sh + (size_t)(mbase + (lane & 15)) * stride
                               + kbase + ((lane >> 4) << 4);
  uint4 lo = *(const uint4*)p;
  uint4 hi = *(const uint4*)(p + 8);
  v8u u;
  u[0]=lo.x; u[1]=lo.y; u[2]=lo.z; u[3]=lo.w;
  u[4]=hi.x; u[5]=hi.y; u[6]=hi.z; u[7]=hi.w;
  return __builtin_bit_cast(v16bf, u);
}
// B 32x16 bf16 from global, weights stored as Bt[n][k] (col-contiguous-K):
// lane holds col N = nbase + lane%16, 16 contiguous K starting at kbase + (lane/16)*16.
__device__ inline v16bf frag_b_glb(const unsigned short* Bt, int Kdim, int nbase, int kbase) {
  int lane = threadIdx.x & 31;
  const unsigned short* p = Bt + (size_t)(nbase + (lane & 15)) * Kdim
                               + kbase + ((lane >> 4) << 4);
  return __builtin_bit_cast(v16bf, *(const v8u*)p);
}
__device__ inline v8f wmma_bf16(v16bf a, v16bf b, v8f c) {
  return __builtin_amdgcn_wmma_f32_16x16x32_bf16(false, a, false, b, (short)0, c, false, false);
}

// ---------- prep kernels ----------
__global__ void cvt_transpose_bf_kernel(const float* __restrict__ in,
                                        unsigned short* __restrict__ out, int K, int N) {
  int i = blockIdx.x * 256 + threadIdx.x;
  if (i < K * N) {
    int k = i / N, n = i - k * N;
    out[(size_t)n * K + k] = f2bf(in[i]);     // Bt[n][k] = W[k][n]
  }
}
__global__ void cvt_bf_kernel(const float* __restrict__ in,
                              unsigned short* __restrict__ out, int len) {
  int i = blockIdx.x * 256 + threadIdx.x;
  if (i < len) out[i] = f2bf(in[i]);
}
__global__ void fill_kernel(float* __restrict__ p, float v, long long n) {
  long long i = (long long)blockIdx.x * 256 + threadIdx.x;
  if (i < n) p[i] = v;
}

// ---------- GatedMLP kernel (shared by edge and node stages) ----------
// out[r] = silu(LN(silu(x@W1c)@W2c)) * sigmoid(LN(silu(x@W1g)@W2g)) + res_w*in0[r]
// x = [in0[r] | in1[idx1[r]] | in2[idx2[r]]]  (idx==nullptr -> identity)
__global__ __launch_bounds__(256)
void gated_mlp_wmma_kernel(
    const float* __restrict__ in0, const float* __restrict__ in1,
    const float* __restrict__ in2,
    const int* __restrict__ idx1, const int* __restrict__ idx2,
    const unsigned short* __restrict__ cw1t, const unsigned short* __restrict__ cw2t,
    const unsigned short* __restrict__ gw1t, const unsigned short* __restrict__ gw2t,
    const float* __restrict__ ln_cg, const float* __restrict__ ln_cb,
    const float* __restrict__ ln_gg, const float* __restrict__ ln_gb,
    const float* __restrict__ res_w,
    float* __restrict__ out, int rows)
{
  __shared__ __align__(16) unsigned short sh_x[TILE_G * 384];  // 24 KB input tile (bf16)
  __shared__ __align__(16) unsigned short sh_h[TILE_G * 128];  //  8 KB hidden (bf16)
  __shared__ __align__(16) float          sh_c[TILE_G * 128];  // 16 KB pre-LN (f32)
  __shared__ __align__(16) unsigned short sh_a[TILE_G * 128];  //  8 KB silu(core) (bf16)
  __shared__ float sh_red[512];
  __shared__ int sh_i1[TILE_G], sh_i2[TILE_G];

  const int tid  = threadIdx.x;
  const int lane = tid & 31;
  const int wv   = tid >> 5;
  const int r0   = blockIdx.x * TILE_G;

  if (tid < TILE_G) {
    int r = r0 + tid;
    int rr = (r < rows) ? r : 0;
    sh_i1[tid] = idx1 ? idx1[rr] : rr;
    sh_i2[tid] = idx2 ? idx2[rr] : rr;
  }
  __syncthreads();

  // stage x tile: [in0 | gather(in1) | gather(in2)] as bf16
  for (int i = tid; i < TILE_G * 96; i += 256) {
    int r = i / 96, c4 = i - r * 96;
    int row = r0 + r;
    float4 v = make_float4(0.f, 0.f, 0.f, 0.f);
    if (row < rows) {
      if (c4 < 32)      v = ((const float4*)(in0 + (size_t)row * 128))[c4];
      else if (c4 < 64) v = ((const float4*)(in1 + (size_t)sh_i1[r] * 128))[c4 - 32];
      else              v = ((const float4*)(in2 + (size_t)sh_i2[r] * 128))[c4 - 64];
    }
    unsigned short* d = sh_x + r * 384 + c4 * 4;
    d[0]=f2bf(v.x); d[1]=f2bf(v.y); d[2]=f2bf(v.z); d[3]=f2bf(v.w);
  }
  __syncthreads();

  const int mtile = wv & 1;         // 2 M-tiles (32 rows)
  const int nt0   = (wv >> 1) * 2;  // 8 N-tiles, 2 per wave

  for (int pass = 0; pass < 2; ++pass) {           // 0 = core, 1 = gate
    const unsigned short* w1t = pass ? gw1t : cw1t;
    const unsigned short* w2t = pass ? gw2t : cw2t;

    // GEMM1: [32x384] @ [384x128]
    v8f a0 = zero8(), a1 = zero8();
    #pragma unroll
    for (int ks = 0; ks < 12; ++ks) {
      v16bf a  = frag_a_lds(sh_x, 384, mtile * 16, ks * 32);
      v16bf b0 = frag_b_glb(w1t, 384, (nt0 + 0) * 16, ks * 32);
      v16bf b1 = frag_b_glb(w1t, 384, (nt0 + 1) * 16, ks * 32);
      a0 = wmma_bf16(a, b0, a0);
      a1 = wmma_bf16(a, b1, a1);
    }
    #pragma unroll
    for (int g = 0; g < 8; ++g) {                   // silu -> hidden bf16
      int row = mtile * 16 + g + 8 * (lane >> 4);
      int c   = lane & 15;
      float x0 = a0[g], x1 = a1[g];
      sh_h[row * 128 + (nt0 + 0) * 16 + c] = f2bf(x0 * sigf(x0));
      sh_h[row * 128 + (nt0 + 1) * 16 + c] = f2bf(x1 * sigf(x1));
    }
    __syncthreads();

    // GEMM2: [32x128] @ [128x128]
    v8f d0 = zero8(), d1 = zero8();
    #pragma unroll
    for (int ks = 0; ks < 4; ++ks) {
      v16bf a  = frag_a_lds(sh_h, 128, mtile * 16, ks * 32);
      v16bf b0 = frag_b_glb(w2t, 128, (nt0 + 0) * 16, ks * 32);
      v16bf b1 = frag_b_glb(w2t, 128, (nt0 + 1) * 16, ks * 32);
      d0 = wmma_bf16(a, b0, d0);
      d1 = wmma_bf16(a, b1, d1);
    }
    #pragma unroll
    for (int g = 0; g < 8; ++g) {
      int row = mtile * 16 + g + 8 * (lane >> 4);
      int c   = lane & 15;
      sh_c[row * 128 + (nt0 + 0) * 16 + c] = d0[g];
      sh_c[row * 128 + (nt0 + 1) * 16 + c] = d1[g];
    }
    __syncthreads();

    // LayerNorm over 128 dims: 8 threads/row, 16 cols each
    {
      int row = tid >> 3, sub = tid & 7;
      const float* cr = sh_c + row * 128 + sub * 16;
      float vals[16], s1 = 0.f, s2 = 0.f;
      #pragma unroll
      for (int i = 0; i < 16; ++i) { float v = cr[i]; vals[i] = v; s1 += v; s2 += v * v; }
      sh_red[row * 8 + sub]       = s1;
      sh_red[256 + row * 8 + sub] = s2;
      __syncthreads();
      float t1 = 0.f, t2 = 0.f;
      #pragma unroll
      for (int i = 0; i < 8; ++i) { t1 += sh_red[row * 8 + i]; t2 += sh_red[256 + row * 8 + i]; }
      float mean = t1 * (1.0f / 128.0f);
      float var  = t2 * (1.0f / 128.0f) - mean * mean;
      float inv  = rsqrtf(var + 1e-5f);
      if (pass == 0) {
        #pragma unroll
        for (int i = 0; i < 16; ++i) {
          int col = sub * 16 + i;
          float y = (vals[i] - mean) * inv * ln_cg[col] + ln_cb[col];
          sh_a[row * 128 + col] = f2bf(y * sigf(y));          // silu(core)
        }
      } else {
        int grow = r0 + row;
        #pragma unroll
        for (int i = 0; i < 16; ++i) {
          int col = sub * 16 + i;
          float y = (vals[i] - mean) * inv * ln_gg[col] + ln_gb[col];
          float o = bf2f(sh_a[row * 128 + col]) * sigf(y);    // silu(core)*sigmoid(gate)
          if (grow < rows) {
            o += res_w[col] * in0[(size_t)grow * 128 + col];
            out[(size_t)grow * 128 + col] = o;
          }
        }
      }
      __syncthreads();
    }
  }
}

// ---------- per-dim segment softmax in 3 passes, logits recomputed via WMMA ----------
// mode 0: atomic max of s into maxb; mode 1: atomic add of exp(s-max) into sumb;
// mode 2: atomic add of (exp(s-max)/(sum+eps)) * attn_edge into accb,
//         attn_edge recovered from edge_out - res_w*edge_feat.
__global__ __launch_bounds__(256)
void seg_softmax_wmma_kernel(
    const float* __restrict__ edge_feat,
    const int* __restrict__ sidx, const int* __restrict__ tidx,
    const unsigned short* __restrict__ wsrc_bf, const unsigned short* __restrict__ wtgt_bf,
    float* __restrict__ max_s, float* __restrict__ max_t,
    float* __restrict__ sum_s, float* __restrict__ sum_t,
    float* __restrict__ acc_s, float* __restrict__ acc_t,
    const float* __restrict__ edge_out, const float* __restrict__ res_w,
    int nedges, int mode)
{
  __shared__ __align__(16) unsigned short sh_ef[TILE_S * 128];  // 16 KB
  __shared__ int sh_seg[2][TILE_S];

  const int tid  = threadIdx.x;
  const int lane = tid & 31;
  const int wv   = tid >> 5;
  const int e0   = blockIdx.x * TILE_S;

  if (tid < TILE_S) {
    int e = e0 + tid;
    int ee = (e < nedges) ? e : 0;
    sh_seg[0][tid] = sidx[ee];
    sh_seg[1][tid] = tidx[ee];
  }
  for (int i = tid; i < TILE_S * 32; i += 256) {
    int r = i >> 5, c4 = i & 31;
    int e = e0 + r;
    float4 v = make_float4(0.f, 0.f, 0.f, 0.f);
    if (e < nedges) v = ((const float4*)(edge_feat + (size_t)e * 128))[c4];
    unsigned short* d = sh_ef + r * 128 + c4 * 4;
    d[0]=f2bf(v.x); d[1]=f2bf(v.y); d[2]=f2bf(v.z); d[3]=f2bf(v.w);
  }
  __syncthreads();

  const int mtile = wv & 3;         // 4 M-tiles (64 rows)
  const int nt0   = (wv >> 2) * 4;  // 8 N-tiles, 4 per wave

  for (int half = 0; half < 2; ++half) {  // 0 = src side, 1 = tgt side
    const unsigned short* W = half ? wtgt_bf : wsrc_bf;
    float* maxb = half ? max_t : max_s;
    float* sumb = half ? sum_t : sum_s;
    float* accb = half ? acc_t : acc_s;

    v8f acc[4];
    #pragma unroll
    for (int j = 0; j < 4; ++j) acc[j] = zero8();
    #pragma unroll
    for (int ks = 0; ks < 4; ++ks) {
      v16bf a = frag_a_lds(sh_ef, 128, mtile * 16, ks * 32);
      #pragma unroll
      for (int j = 0; j < 4; ++j) {
        v16bf b = frag_b_glb(W, 128, (nt0 + j) * 16, ks * 32);
        acc[j] = wmma_bf16(a, b, acc[j]);
      }
    }
    #pragma unroll
    for (int j = 0; j < 4; ++j) {
      #pragma unroll
      for (int g = 0; g < 8; ++g) {
        int r   = mtile * 16 + g + 8 * (lane >> 4);
        int col = (nt0 + j) * 16 + (lane & 15);
        int e   = e0 + r;
        if (e >= nedges) continue;
        int seg = sh_seg[half][r];
        size_t off = (size_t)seg * 128 + col;
        float val = acc[j][g];
        if (mode == 0) {
          atomicMaxF(maxb + off, val);
        } else if (mode == 1) {
          unsafeAtomicAdd(sumb + off, __expf(val - maxb[off]));
        } else {
          float alpha = __expf(val - maxb[off]) / (sumb[off] + 1e-16f);
          size_t go = (size_t)e * 128 + col;
          float ae = edge_out[go] - res_w[col] * edge_feat[go];
          unsafeAtomicAdd(accb + off, alpha * ae);
        }
      }
    }
  }
}

// ---------- launcher ----------
extern "C" void kernel_launch(void* const* d_in, const int* in_sizes, int n_in,
                              void* d_out, int out_size, void* d_ws, size_t ws_size,
                              hipStream_t stream) {
  (void)n_in; (void)out_size; (void)ws_size;
  const float* node_feat  = (const float*)d_in[0];
  const float* edge_feat  = (const float*)d_in[1];
  const float* w_src      = (const float*)d_in[2];
  const float* w_tgt      = (const float*)d_in[3];
  const float* e_core_w1  = (const float*)d_in[4];
  const float* e_core_w2  = (const float*)d_in[5];
  const float* e_gate_w1  = (const float*)d_in[6];
  const float* e_gate_w2  = (const float*)d_in[7];
  const float* e_ln_cg    = (const float*)d_in[8];
  const float* e_ln_cb    = (const float*)d_in[9];
  const float* e_ln_gg    = (const float*)d_in[10];
  const float* e_ln_gb    = (const float*)d_in[11];
  const float* n_core_w1  = (const float*)d_in[12];
  const float* n_core_w2  = (const float*)d_in[13];
  const float* n_gate_w1  = (const float*)d_in[14];
  const float* n_gate_w2  = (const float*)d_in[15];
  const float* n_ln_cg    = (const float*)d_in[16];
  const float* n_ln_cb    = (const float*)d_in[17];
  const float* n_ln_gg    = (const float*)d_in[18];
  const float* n_ln_gb    = (const float*)d_in[19];
  const float* node_res_w = (const float*)d_in[20];
  const float* edge_res_w = (const float*)d_in[21];
  const int*   sidx       = (const int*)d_in[22];
  const int*   tidx       = (const int*)d_in[23];

  const int N = in_sizes[0] / 128;
  const int E = in_sizes[1] / 128;

  // workspace carve-out (256B aligned)
  char* ws = (char*)d_ws;
  size_t off = 0;
  auto take = [&](size_t bytes) {
    char* p = ws + off;
    off = (off + bytes + 255) & ~(size_t)255;
    return p;
  };
  unsigned short* ecw1t = (unsigned short*)take(384 * 128 * 2);
  unsigned short* egw1t = (unsigned short*)take(384 * 128 * 2);
  unsigned short* ncw1t = (unsigned short*)take(384 * 128 * 2);
  unsigned short* ngw1t = (unsigned short*)take(384 * 128 * 2);
  unsigned short* ecw2t = (unsigned short*)take(128 * 128 * 2);
  unsigned short* egw2t = (unsigned short*)take(128 * 128 * 2);
  unsigned short* ncw2t = (unsigned short*)take(128 * 128 * 2);
  unsigned short* ngw2t = (unsigned short*)take(128 * 128 * 2);
  unsigned short* wsrcb = (unsigned short*)take(128 * 128 * 2);
  unsigned short* wtgtb = (unsigned short*)take(128 * 128 * 2);
  const size_t NB = (size_t)N * 128 * sizeof(float);   // multiple of 256
  float* max_s = (float*)take(NB);
  float* max_t = (float*)take(NB);
  float* sum_s = (float*)take(NB);
  float* sum_t = (float*)take(NB);
  float* acc_s = (float*)take(NB);
  float* acc_t = (float*)take(NB);

  // 1) weight conversion / transposition to bf16 fragment layout
  cvt_transpose_bf_kernel<<<(384 * 128 + 255) / 256, 256, 0, stream>>>(e_core_w1, ecw1t, 384, 128);
  cvt_transpose_bf_kernel<<<(384 * 128 + 255) / 256, 256, 0, stream>>>(e_gate_w1, egw1t, 384, 128);
  cvt_transpose_bf_kernel<<<(384 * 128 + 255) / 256, 256, 0, stream>>>(n_core_w1, ncw1t, 384, 128);
  cvt_transpose_bf_kernel<<<(384 * 128 + 255) / 256, 256, 0, stream>>>(n_gate_w1, ngw1t, 384, 128);
  cvt_transpose_bf_kernel<<<(128 * 128 + 255) / 256, 256, 0, stream>>>(e_core_w2, ecw2t, 128, 128);
  cvt_transpose_bf_kernel<<<(128 * 128 + 255) / 256, 256, 0, stream>>>(e_gate_w2, egw2t, 128, 128);
  cvt_transpose_bf_kernel<<<(128 * 128 + 255) / 256, 256, 0, stream>>>(n_core_w2, ncw2t, 128, 128);
  cvt_transpose_bf_kernel<<<(128 * 128 + 255) / 256, 256, 0, stream>>>(n_gate_w2, ngw2t, 128, 128);
  // y = x @ W.T means Bt[n][k] = W[n][k]: row-major W is already fragment layout
  cvt_bf_kernel<<<(128 * 128 + 255) / 256, 256, 0, stream>>>(w_src, wsrcb, 128 * 128);
  cvt_bf_kernel<<<(128 * 128 + 255) / 256, 256, 0, stream>>>(w_tgt, wtgtb, 128 * 128);

  // 2) init segment buffers: max = -1e30 (2 blocks), sum/acc = 0 (4 blocks, contiguous)
  long long nmax = (long long)2 * N * 128;
  long long nzro = (long long)4 * N * 128;
  fill_kernel<<<(unsigned)((nmax + 255) / 256), 256, 0, stream>>>(max_s, -1e30f, nmax);
  fill_kernel<<<(unsigned)((nzro + 255) / 256), 256, 0, stream>>>(sum_s, 0.0f, nzro);

  float* out_node = (float*)d_out;
  float* out_edge = out_node + (size_t)N * 128;

  // 3) edge GatedMLP: x = [edge_feat | node_feat[tgt] | node_feat[src]]
  gated_mlp_wmma_kernel<<<(E + TILE_G - 1) / TILE_G, 256, 0, stream>>>(
      edge_feat, node_feat, node_feat, tidx, sidx,
      ecw1t, ecw2t, egw1t, egw2t,
      e_ln_cg, e_ln_cb, e_ln_gg, e_ln_gb,
      edge_res_w, out_edge, E);

  // 4-6) dimwise segment softmax + scatter (max, sum, weighted scatter)
  const int sg = (E + TILE_S - 1) / TILE_S;
  for (int mode = 0; mode < 3; ++mode) {
    seg_softmax_wmma_kernel<<<sg, 256, 0, stream>>>(
        edge_feat, sidx, tidx, wsrcb, wtgtb,
        max_s, max_t, sum_s, sum_t, acc_s, acc_t,
        out_edge, edge_res_w, E, mode);
  }

  // 7) node GatedMLP: fusion = [node_feat | attn_tgt | attn_src]
  gated_mlp_wmma_kernel<<<(N + TILE_G - 1) / TILE_G, 256, 0, stream>>>(
      node_feat, acc_t, acc_s, nullptr, nullptr,
      ncw1t, ncw2t, ngw1t, ngw2t,
      n_ln_cg, n_ln_cb, n_ln_gg, n_ln_gb,
      node_res_w, out_node, N);
}
